// GATNet_21165598834943
// MI455X (gfx1250) — compile-verified, matrix operand order checked by reading, not run on hardware
//
#include <hip/hip_runtime.h>

#define GAT_HEADS 4
#define GAT_WIDTH 256
#define NEG_SLOPE 0.2f

typedef __attribute__((ext_vector_type(16))) __bf16 v16bf;
typedef __attribute__((ext_vector_type(8)))  float  v8f;

static __device__ __forceinline__ unsigned short f32_to_bf16(float f) {
  unsigned int u = __float_as_uint(f);
  u += 0x7FFFu + ((u >> 16) & 1u);           // round-to-nearest-even
  return (unsigned short)(u >> 16);
}
// WMMA fragment K-permutation within each 32-chunk: [0..7,16..23,8..15,24..31]
static __device__ __forceinline__ int kperm(int j) {   // j in [0,32)
  return (j >= 8 && j < 24) ? (j ^ 24) : j;
}
// monotone float <-> uint mapping so unsigned atomicMax == float max
static __device__ __forceinline__ unsigned int fmap(float f) {
  unsigned int u = __float_as_uint(f);
  return (u & 0x80000000u) ? ~u : (u | 0x80000000u);
}
static __device__ __forceinline__ float funmap(unsigned int u) {
  unsigned int b = (u & 0x80000000u) ? (u & 0x7FFFFFFFu) : ~u;
  return __uint_as_float(b);
}

__global__ void k_zero32(unsigned int* p, size_t n) {
  size_t i = (size_t)blockIdx.x * blockDim.x + threadIdx.x;
  size_t stride = (size_t)gridDim.x * blockDim.x;
  for (; i < n; i += stride) p[i] = 0u;
}

// pack activations [M x K] f32 -> bf16, K-permuted per 32-chunk (row-major)
__global__ void k_pack_a(const float* __restrict__ src,
                         unsigned short* __restrict__ dst, size_t n, int K) {
  size_t i = (size_t)blockIdx.x * blockDim.x + threadIdx.x;
  size_t stride = (size_t)gridDim.x * blockDim.x;
  for (; i < n; i += stride) {
    size_t row = i / K;
    int k = (int)(i - row * K);
    int pk = (k & ~31) | kperm(k & 31);
    dst[row * K + pk] = f32_to_bf16(src[i]);
  }
}

// pack W [K x 256] f32 -> bf16, transposed to [col][K] with K-permutation
__global__ void k_pack_b(const float* __restrict__ src,
                         unsigned short* __restrict__ dst, size_t n, int K) {
  size_t i = (size_t)blockIdx.x * blockDim.x + threadIdx.x;
  size_t stride = (size_t)gridDim.x * blockDim.x;
  for (; i < n; i += stride) {
    int k   = (int)(i >> 8);          // row in W
    int col = (int)(i & 255);
    int pk = (k & ~31) | kperm(k & 31);
    dst[(size_t)col * K + pk] = f32_to_bf16(src[i]);
  }
}

// C[M x 256] = A[M x K] * W[K x 256]; operands pre-packed in fragment order.
// grid = (M/16, 2); block = 256 (8 waves); wave w owns cols n0 + w*16 .. +15.
// Inner loop: 2x 32B contiguous loads + one v_wmma_f32_16x16x32_bf16.
__global__ __launch_bounds__(256)
void k_gemm_wmma(const unsigned short* __restrict__ Apack,
                 const unsigned short* __restrict__ Bpack,
                 float* __restrict__ C, int K) {
  const int tid  = threadIdx.x;
  const int wave = tid >> 5;
  const int lane = tid & 31;
  const int l16  = lane & 15;
  const bool hi  = lane >= 16;
  const int mtile = blockIdx.x << 4;
  const int n0    = blockIdx.y << 7;
  const int ncol  = n0 + (wave << 4) + l16;
  const int m     = mtile + l16;
  const int ksel  = hi ? 16 : 0;     // lanes 0-15: perm pos 0..15; 16-31: 16..31

  const unsigned short* ap = Apack + (size_t)m    * K + ksel;
  const unsigned short* bp = Bpack + (size_t)ncol * K + ksel;

  v8f acc = {};
  for (int k0 = 0; k0 < K; k0 += 32) {
    if (k0 + 64 < K) {               // prefetch two chunks ahead
      __builtin_prefetch(ap + k0 + 64, 0, 1);
      __builtin_prefetch(bp + k0 + 64, 0, 1);
    }
    v16bf a = *(const v16bf*)(ap + k0);
    v16bf b = *(const v16bf*)(bp + k0);
    acc = __builtin_amdgcn_wmma_f32_16x16x32_bf16(false, a, false, b,
                                                  (short)0, acc, false, false);
  }
  // C/D layout: reg r -> M = r + 8*(lane>=16), N = lane%16
  const int rowbase = mtile + (hi ? 8 : 0);
#pragma unroll
  for (int r = 0; r < 8; ++r)
    C[(size_t)(rowbase + r) * GAT_WIDTH + ncol] = acc[r];
}

// per-node attention logits: al[n,h] = sum_c h[n,h*64+c]*a[h,c]
__global__ __launch_bounds__(256)
void k_attn(const float* __restrict__ hfeat,
            const float* __restrict__ a_src, const float* __restrict__ a_dst,
            float* __restrict__ als, float* __restrict__ ald) {
  __shared__ float ss[256], sd[256];
  const int n = blockIdx.x, t = threadIdx.x;
  float hv = hfeat[(size_t)n * GAT_WIDTH + t];
  ss[t] = hv * a_src[t];
  sd[t] = hv * a_dst[t];
  __syncthreads();
  const int c = t & 63;
  for (int off = 32; off > 0; off >>= 1) {
    if (c < off) { ss[t] += ss[t + off]; sd[t] += sd[t + off]; }
    __syncthreads();
  }
  if (c == 0) {
    int head = t >> 6;
    als[(size_t)n * GAT_HEADS + head] = ss[t];
    ald[(size_t)n * GAT_HEADS + head] = sd[t];
  }
}

__global__ void k_edge_max(const int* __restrict__ ei,
                           const float* __restrict__ als, const float* __restrict__ ald,
                           float* __restrict__ elog, unsigned int* __restrict__ m,
                           int E, int ET) {
  int e = blockIdx.x * blockDim.x + threadIdx.x;
  if (e >= ET) return;
  int s, d;
  if (e < E) { s = ei[e]; d = ei[E + e]; } else { s = e - E; d = s; }
#pragma unroll
  for (int h = 0; h < GAT_HEADS; ++h) {
    float v = als[(size_t)s * GAT_HEADS + h] + ald[(size_t)d * GAT_HEADS + h];
    v = v > 0.0f ? v : NEG_SLOPE * v;            // leaky relu
    elog[(size_t)e * GAT_HEADS + h] = v;
    atomicMax(&m[(size_t)d * GAT_HEADS + h], fmap(v));
  }
}

__global__ void k_edge_exp(const int* __restrict__ ei,
                           float* __restrict__ elog,
                           const unsigned int* __restrict__ m,
                           float* __restrict__ denom, int E, int ET) {
  int e = blockIdx.x * blockDim.x + threadIdx.x;
  if (e >= ET) return;
  int d = (e < E) ? ei[E + e] : (e - E);
#pragma unroll
  for (int h = 0; h < GAT_HEADS; ++h) {
    float mm = funmap(m[(size_t)d * GAT_HEADS + h]);
    float w = expf(elog[(size_t)e * GAT_HEADS + h] - mm);
    elog[(size_t)e * GAT_HEADS + h] = w;
    atomicAdd(&denom[(size_t)d * GAT_HEADS + h], w);
  }
}

// one 256-thread block per edge: out[dst, :] += h[src, :] * alpha[head]
__global__ __launch_bounds__(256)
void k_edge_msg(const int* __restrict__ ei, const float* __restrict__ elog,
                const float* __restrict__ denom, const float* __restrict__ hfeat,
                float* __restrict__ hagg, int E) {
  const int e = blockIdx.x, t = threadIdx.x, h = t >> 6;
  int s, d;
  if (e < E) { s = ei[e]; d = ei[E + e]; } else { s = e - E; d = s; }
  float w = elog[(size_t)e * GAT_HEADS + h];
  float alpha = w / (denom[(size_t)d * GAT_HEADS + h] + 1e-16f);
  atomicAdd(&hagg[(size_t)d * GAT_WIDTH + t],
            hfeat[(size_t)s * GAT_WIDTH + t] * alpha);
}

__global__ void k_elu_bias(float* __restrict__ hv, const float* __restrict__ bias, size_t n) {
  size_t i = (size_t)blockIdx.x * blockDim.x + threadIdx.x;
  size_t stride = (size_t)gridDim.x * blockDim.x;
  for (; i < n; i += stride) {
    float v = hv[i] + bias[i & (GAT_WIDTH - 1)];
    hv[i] = v > 0.0f ? v : (expf(v) - 1.0f);     // elu, alpha=1
  }
}

__global__ __launch_bounds__(256)
void k_pool(const float* __restrict__ hv, const int* __restrict__ batch,
            float* __restrict__ pooled) {
  const int n = blockIdx.x, c = threadIdx.x;
  atomicAdd(&pooled[(size_t)batch[n] * GAT_WIDTH + c],
            hv[(size_t)n * GAT_WIDTH + c]);
}

__global__ __launch_bounds__(256)
void k_final(const float* __restrict__ pooled, const float* __restrict__ lin_w,
             const float* __restrict__ lin_b, float* __restrict__ out) {
  __shared__ float s[256];
  const int g = blockIdx.x, t = threadIdx.x;
  s[t] = pooled[(size_t)g * GAT_WIDTH + t] * lin_w[t];
  __syncthreads();
  for (int off = 128; off > 0; off >>= 1) {
    if (t < off) s[t] += s[t + off];
    __syncthreads();
  }
  if (t == 0) out[g] = s[0] + lin_b[0];
}

extern "C" void kernel_launch(void* const* d_in, const int* in_sizes, int n_in,
                              void* d_out, int out_size, void* d_ws, size_t ws_size,
                              hipStream_t stream) {
  const float* x      = (const float*)d_in[0];
  const int*   ei     = (const int*)d_in[1];
  // d_in[2] = edge_attr (unused by reference)
  const int*   batch  = (const int*)d_in[3];
  const float* W[3]   = {(const float*)d_in[4],  (const float*)d_in[8],  (const float*)d_in[12]};
  const float* asr[3] = {(const float*)d_in[5],  (const float*)d_in[9],  (const float*)d_in[13]};
  const float* adt[3] = {(const float*)d_in[6],  (const float*)d_in[10], (const float*)d_in[14]};
  const float* bia[3] = {(const float*)d_in[7],  (const float*)d_in[11], (const float*)d_in[15]};
  const float* lin_w  = (const float*)d_in[16];
  const float* lin_b  = (const float*)d_in[17];

  const int IN_DIM = 128;
  const int N  = in_sizes[0] / IN_DIM;
  const int E  = in_sizes[1] / 2;
  const int ET = E + N;
  const int G  = out_size;

  // ---- workspace carve (~139 MB) ----
  char* wp = (char*)d_ws;
  auto take = [&](size_t bytes) -> void* {
    void* r = (void*)wp;
    wp += (bytes + 255) & ~(size_t)255;
    return r;
  };
  float*          hcur  = (float*)take((size_t)N * GAT_WIDTH * 4);   // agg target, then next input
  float*          hfeat = (float*)take((size_t)N * GAT_WIDTH * 4);   // x@W per layer
  unsigned short* xb    = (unsigned short*)take((size_t)N * GAT_WIDTH * 2);
  unsigned short* Wb    = (unsigned short*)take((size_t)GAT_WIDTH * GAT_WIDTH * 2);
  float*          als   = (float*)take((size_t)N * GAT_HEADS * 4);
  float*          ald   = (float*)take((size_t)N * GAT_HEADS * 4);
  unsigned int*   mbuf  = (unsigned int*)take((size_t)N * GAT_HEADS * 4);
  float*          denom = (float*)take((size_t)N * GAT_HEADS * 4);
  float*          elog  = (float*)take((size_t)ET * GAT_HEADS * 4);
  float*          pooled= (float*)take((size_t)G * GAT_WIDTH * 4);
  (void)ws_size; (void)n_in;

  const float* cur = x;
  int K = IN_DIM;
  const int eblk = (ET + 255) / 256;
  for (int l = 0; l < 3; ++l) {
    k_pack_a<<<1024, 256, 0, stream>>>(cur, xb, (size_t)N * K, K);
    k_pack_b<<<256, 256, 0, stream>>>(W[l], Wb, (size_t)K * GAT_WIDTH, K);
    dim3 ggrid(N / 16, GAT_WIDTH / 128);
    k_gemm_wmma<<<ggrid, 256, 0, stream>>>(xb, Wb, hfeat, K);
    k_attn<<<N, 256, 0, stream>>>(hfeat, asr[l], adt[l], als, ald);

    k_zero32<<<1024, 256, 0, stream>>>((unsigned int*)hcur, (size_t)N * GAT_WIDTH);
    k_zero32<<<256, 256, 0, stream>>>(mbuf, (size_t)N * GAT_HEADS);
    k_zero32<<<256, 256, 0, stream>>>((unsigned int*)denom, (size_t)N * GAT_HEADS);

    k_edge_max<<<eblk, 256, 0, stream>>>(ei, als, ald, elog, mbuf, E, ET);
    k_edge_exp<<<eblk, 256, 0, stream>>>(ei, elog, mbuf, denom, E, ET);
    k_edge_msg<<<ET, 256, 0, stream>>>(ei, elog, denom, hfeat, hcur, E);
    k_elu_bias<<<1024, 256, 0, stream>>>(hcur, bia[l], (size_t)N * GAT_WIDTH);

    cur = hcur;
    K = GAT_WIDTH;
  }

  k_zero32<<<256, 256, 0, stream>>>((unsigned int*)pooled, (size_t)G * GAT_WIDTH);
  k_pool<<<N, 256, 0, stream>>>(hcur, batch, pooled);
  k_final<<<G, 256, 0, stream>>>(pooled, lin_w, lin_b, (float*)d_out);
}
